// GFBlock_2473901162873
// MI455X (gfx1250) — matrix operand order repaired and checked by reading, not compiled, not run on hardware
//
#include <hip/hip_runtime.h>
#include <math.h>

typedef __attribute__((ext_vector_type(16))) _Float16 v16h;
typedef __attribute__((ext_vector_type(8)))  float    v8f;

#define DEV __device__ __forceinline__

// A-fragment k index for element e (0..15), lane-half h: contiguous k = 16h + e (this code's order; B reads match)
DEV int kmapA(int e, int h) { return 16*h + e; }

// window(flat win, token t) -> flat image row, for img2win with window (wd,wh,ww)
DEV long win_row(long win, int t, int D, int H, int W, int wd, int wh, int ww) {
  int nwX = W / ww, nwY = H / wh, nwZ = D / wd;
  int xw = (int)(win % nwX); long r1 = win / nwX;
  int yw = (int)(r1 % nwY);  long r2 = r1 / nwY;
  int zw = (int)(r2 % nwZ);  long b  = r2 / nwZ;
  int tx = t % ww, ty = (t / ww) % wh, tz = t / (ww * wh);
  long z = (long)zw * wd + tz, y = (long)yw * wh + ty, x = (long)xw * ww + tx;
  return ((b * D + z) * H + y) * W + x;
}

// ---------------- LayerNorm: one wave32 per token (C=256, 8 ch/lane) -------
__global__ void __launch_bounds__(256)
ln_kernel(const float* __restrict__ x, const float* __restrict__ g,
          const float* __restrict__ b, float* __restrict__ y, long nrow) {
  long row = ((long)blockIdx.x * 256 + threadIdx.x) >> 5;
  if (row >= nrow) return;
  int lane = threadIdx.x & 31;
  const float* xr = x + row * 256;
  float v[8]; float s = 0.f;
#pragma unroll
  for (int i = 0; i < 8; ++i) { v[i] = xr[lane * 8 + i]; s += v[i]; }
#pragma unroll
  for (int off = 1; off < 32; off <<= 1) s += __shfl_xor(s, off, 32);
  float mean = s * (1.f / 256.f);
  float var = 0.f;
#pragma unroll
  for (int i = 0; i < 8; ++i) { float d = v[i] - mean; var += d * d; }
#pragma unroll
  for (int off = 1; off < 32; off <<= 1) var += __shfl_xor(var, off, 32);
  float inv = rsqrtf(var * (1.f / 256.f) + 1e-5f);
#pragma unroll
  for (int i = 0; i < 8; ++i) {
    int c = lane * 8 + i;
    y[row * 256 + c] = (v[i] - mean) * inv * g[c] + b[c];
  }
}

// ---------------- Generic WMMA GEMM: out[R,M] = A[R,K] * Wt[K,M] + bias ----
// R % 64 == 0, K % 32 == 0, M % 16 == 0 (M guarded per 16-col group).
// fp32 in/out, f16 WMMA, fp32 accumulate. B staged TRANSPOSED so fragment
// reads are contiguous ds_load_b128s.
__global__ void __launch_bounds__(256)
gemm_wmma(const float* __restrict__ A, const float* __restrict__ Wt,
          const float* __restrict__ bias, float* __restrict__ out,
          int R, int K, int M) {
  __shared__ _Float16 Ah[64][40];    // [row][k]
  __shared__ _Float16 Bt[128][40];   // [col][k]  (transposed)
  int tid = threadIdx.x, wave = tid >> 5, lane = tid & 31;
  int half = lane >> 4, m = lane & 15;
  int wy = wave >> 2, wx = wave & 3;
  long rblk = (long)blockIdx.y * 64;
  int  cblk = blockIdx.x * 128;
  v8f acc[2][2] = {};
  for (int kk = 0; kk < K; kk += 32) {
    { // stage A tile 64x32 : 2x float4 per thread, contiguous f16 row writes
      int r = tid >> 2, c0 = (tid & 3) * 8;
      const float* src = A + (rblk + r) * (long)K + kk + c0;
      if (kk + 32 < K) __builtin_prefetch(src + 32, 0, 3);
      float4 f0 = *(const float4*)(src);
      float4 f1 = *(const float4*)(src + 4);
      Ah[r][c0 + 0] = (_Float16)f0.x; Ah[r][c0 + 1] = (_Float16)f0.y;
      Ah[r][c0 + 2] = (_Float16)f0.z; Ah[r][c0 + 3] = (_Float16)f0.w;
      Ah[r][c0 + 4] = (_Float16)f1.x; Ah[r][c0 + 5] = (_Float16)f1.y;
      Ah[r][c0 + 6] = (_Float16)f1.z; Ah[r][c0 + 7] = (_Float16)f1.w;
    }
    { // stage B tile 32x128 transposed; per-thread-uniform M guard (M%16==0)
      int r = tid >> 3, c0 = (tid & 7) * 16;
      if (cblk + c0 + 16 <= M) {
        const float* src = Wt + (long)(kk + r) * M + cblk + c0;
        float4 f0 = *(const float4*)(src);
        float4 f1 = *(const float4*)(src + 4);
        float4 f2 = *(const float4*)(src + 8);
        float4 f3 = *(const float4*)(src + 12);
        Bt[c0 +  0][r] = (_Float16)f0.x; Bt[c0 +  1][r] = (_Float16)f0.y;
        Bt[c0 +  2][r] = (_Float16)f0.z; Bt[c0 +  3][r] = (_Float16)f0.w;
        Bt[c0 +  4][r] = (_Float16)f1.x; Bt[c0 +  5][r] = (_Float16)f1.y;
        Bt[c0 +  6][r] = (_Float16)f1.z; Bt[c0 +  7][r] = (_Float16)f1.w;
        Bt[c0 +  8][r] = (_Float16)f2.x; Bt[c0 +  9][r] = (_Float16)f2.y;
        Bt[c0 + 10][r] = (_Float16)f2.z; Bt[c0 + 11][r] = (_Float16)f2.w;
        Bt[c0 + 12][r] = (_Float16)f3.x; Bt[c0 + 13][r] = (_Float16)f3.y;
        Bt[c0 + 14][r] = (_Float16)f3.z; Bt[c0 + 15][r] = (_Float16)f3.w;
      } else {
#pragma unroll
        for (int j = 0; j < 16; ++j) Bt[c0 + j][r] = (_Float16)0.f;
      }
    }
    __syncthreads();
    v16h af[2], bf[2];
#pragma unroll
    for (int t = 0; t < 2; ++t) {
      int rbase = wy * 32 + t * 16;
#pragma unroll
      for (int e = 0; e < 16; ++e) af[t][e] = Ah[rbase + m][kmapA(e, half)];
      int cbase = wx * 32 + t * 16;
#pragma unroll
      for (int e = 0; e < 16; ++e) bf[t][e] = Bt[cbase + m][16 * half + e];
    }
#pragma unroll
    for (int a = 0; a < 2; ++a)
#pragma unroll
      for (int b2 = 0; b2 < 2; ++b2)
        acc[a][b2] = __builtin_amdgcn_wmma_f32_16x16x32_f16(
            false, af[a], false, bf[b2], (short)0, acc[a][b2], false, false);
    __syncthreads();
  }
#pragma unroll
  for (int a = 0; a < 2; ++a)
#pragma unroll
    for (int b2 = 0; b2 < 2; ++b2)
#pragma unroll
      for (int i = 0; i < 8; ++i) {
        long r = rblk + wy * 32 + a * 16 + i + 8 * half;
        int  c = cblk + wx * 32 + b2 * 16 + m;
        if (c < M && r < R)
          out[r * (long)M + c] = acc[a][b2][i] + (bias ? bias[c] : 0.f);
      }
}

// ---------------- Flash attention over one (window, head) per block -------
// HD = head dim (8 zero-padded to K=32, or 32 exact). 8 waves, 16 q rows each.
// K staged [tok][ch], V staged transposed [ch][tok]; both zero-padded to 32
// channels so fragment reads are unconditional contiguous b128s.
template <int HD>
__global__ void __launch_bounds__(256)
attn_flash(const float* __restrict__ qp, const float* __restrict__ kp,
           const float* __restrict__ vp, int ld, int cs, float scale,
           int D, int H, int W, int wd, int wh, int ww, int L,
           float* __restrict__ obuf, int Cb) {
  __shared__ _Float16 Kl[32][40];        // [token][channel] (pad to 32 ch)
  __shared__ _Float16 Vt[32][40];        // [channel][token] (rows >=HD zero)
  __shared__ _Float16 Pst[8][16][40];
  int tid = threadIdx.x, wave = tid >> 5, lane = tid & 31;
  int half = lane >> 4, m = lane & 15;
  long win = blockIdx.x; int head = blockIdx.y;
  int chunks = L / 32;
  int stok = tid >> 3, sc0 = (tid & 7) * 4;    // staging: 8 thr/token, 4 ch
  for (int r0 = wave * 16; r0 < L; r0 += 128) {
    v16h qa;
    long grow = win_row(win, r0 + m, D, H, W, wd, wh, ww);
    const float* qrow = qp + grow * ld + (long)cs * head * HD;
#pragma unroll
    for (int e = 0; e < 16; ++e) {
      int k = kmapA(e, half);
      float v = scale * qrow[(long)cs * k];  // stays within token row even k>=HD
      if (HD < 32) v = (k < HD) ? v : 0.f;
      qa[e] = (_Float16)v;
    }
    float mrow[8], lrow[8];
    v8f acc0 = {}, acc1 = {};
#pragma unroll
    for (int i = 0; i < 8; ++i) { mrow[i] = -1e30f; lrow[i] = 0.f; }
    for (int kc = 0; kc < chunks; ++kc) {
      { // stage K,V chunk: one win_row per thread
        long gr = win_row(win, kc * 32 + stok, D, H, W, wd, wh, ww);
        const float* kb = kp + gr * ld + (long)cs * head * HD;
        const float* vb = vp + gr * ld + (long)cs * head * HD;
#pragma unroll
        for (int i = 0; i < 4; ++i) {
          int c = sc0 + i;
          float kf = 0.f, vf = 0.f;
          if (c < HD) { kf = kb[(long)cs * c]; vf = vb[(long)cs * c]; }
          Kl[stok][c] = (_Float16)kf;
          Vt[c][stok] = (_Float16)vf;
        }
      }
      __syncthreads();
      v16h kb0, kb1;
#pragma unroll
      for (int e = 0; e < 16; ++e) {
        kb0[e] = Kl[m][16 * half + e];
        kb1[e] = Kl[16 + m][16 * half + e];
      }
      v8f z0 = {}, z1 = {};
      v8f S0 = __builtin_amdgcn_wmma_f32_16x16x32_f16(false, qa, false, kb0, (short)0, z0, false, false);
      v8f S1 = __builtin_amdgcn_wmma_f32_16x16x32_f16(false, qa, false, kb1, (short)0, z1, false, false);
#pragma unroll
      for (int i = 0; i < 8; ++i) {                       // online softmax
        float s = fmaxf(S0[i], S1[i]);
#pragma unroll
        for (int off = 1; off < 16; off <<= 1) s = fmaxf(s, __shfl_xor(s, off, 32));
        float mn = fmaxf(mrow[i], s);
        float alpha = __expf(mrow[i] - mn);
        mrow[i] = mn;
        float p0 = __expf(S0[i] - mn), p1 = __expf(S1[i] - mn);
        S0[i] = p0; S1[i] = p1;
        float r = p0 + p1;
#pragma unroll
        for (int off = 1; off < 16; off <<= 1) r += __shfl_xor(r, off, 32);
        lrow[i] = lrow[i] * alpha + r;
        acc0[i] *= alpha; acc1[i] *= alpha;
      }
#pragma unroll
      for (int i = 0; i < 8; ++i) {                       // P (C-layout) -> LDS
        Pst[wave][i + 8 * half][m]      = (_Float16)S0[i];
        Pst[wave][i + 8 * half][16 + m] = (_Float16)S1[i];
      }
      v16h pa;
#pragma unroll
      for (int e = 0; e < 16; ++e) pa[e] = Pst[wave][m][kmapA(e, half)];
      v16h vb0;
#pragma unroll
      for (int e = 0; e < 16; ++e) vb0[e] = Vt[m][16 * half + e];
      acc0 = __builtin_amdgcn_wmma_f32_16x16x32_f16(false, pa, false, vb0, (short)0, acc0, false, false);
      if constexpr (HD > 16) {
        v16h vb1;
#pragma unroll
        for (int e = 0; e < 16; ++e) vb1[e] = Vt[16 + m][16 * half + e];
        acc1 = __builtin_amdgcn_wmma_f32_16x16x32_f16(false, pa, false, vb1, (short)0, acc1, false, false);
      }
      __syncthreads();
    }
#pragma unroll
    for (int i = 0; i < 8; ++i) {                         // epilogue O /= l
      int t = r0 + i + 8 * half;
      long orow = win * L + t;
      float inv = 1.f / lrow[i];
      if (m < HD) obuf[orow * Cb + head * HD + m] = acc0[i] * inv;
      if constexpr (HD > 16) obuf[orow * Cb + head * HD + 16 + m] = acc1[i] * inv;
    }
  }
}

// ---------------- Tiny-window attention: ws=(4,1,1), L=4, HD=32 -----------
__global__ void __launch_bounds__(256)
attn_small_k(const float* __restrict__ qp, const float* __restrict__ kp,
             const float* __restrict__ vp, int ld, int cs, float scale,
             int D, int H, int W, float* __restrict__ obuf, long total) {
  long idx = (long)blockIdx.x * 256 + threadIdx.x;
  if (idx >= total) return;
  int qt = (int)(idx & 3); int head = (int)((idx >> 2) & 7); long win = idx >> 5;
  long gr[4];
#pragma unroll
  for (int j = 0; j < 4; ++j) gr[j] = win_row(win, j, D, H, W, 4, 1, 1);
  float s[4] = {0.f, 0.f, 0.f, 0.f};
  for (int c = 0; c < 32; ++c) {
    float qv = qp[gr[qt] * ld + (long)cs * (head * 32 + c)];
#pragma unroll
    for (int j = 0; j < 4; ++j) s[j] += qv * kp[gr[j] * ld + (long)cs * (head * 32 + c)];
  }
  float mx = fmaxf(fmaxf(s[0], s[1]), fmaxf(s[2], s[3]));
  float p[4], l = 0.f;
#pragma unroll
  for (int j = 0; j < 4; ++j) { p[j] = __expf(scale * (s[j] - mx)); l += p[j]; }
  float inv = 1.f / l;
  for (int c = 0; c < 32; ++c) {
    float o = 0.f;
#pragma unroll
    for (int j = 0; j < 4; ++j) o += p[j] * vp[gr[j] * ld + (long)cs * (head * 32 + c)];
    obuf[(win * 4 + qt) * 256 + head * 32 + c] = o * inv;
  }
}

// ---------------- Depthwise 3x3x3 conv (zero pad), strided input ----------
__global__ void __launch_bounds__(256)
dwconv3d_k(const float* __restrict__ in, int ld, int cs,
           const float* __restrict__ w, const float* __restrict__ bias,
           float* __restrict__ out, int D, int H, int W, int Cc, long total) {
  long idx = (long)blockIdx.x * 256 + threadIdx.x;
  if (idx >= total) return;
  int c = (int)(idx % Cc); long r = idx / Cc;
  int x = (int)(r % W); int y = (int)((r / W) % H); int z = (int)(r / ((long)W * H));
  float s = bias[c];
  for (int dz = -1; dz <= 1; ++dz) {
    int zz = z + dz; if (zz < 0 || zz >= D) continue;
    for (int dy = -1; dy <= 1; ++dy) {
      int yy = y + dy; if (yy < 0 || yy >= H) continue;
      for (int dx = -1; dx <= 1; ++dx) {
        int xx = x + dx; if (xx < 0 || xx >= W) continue;
        long rr = ((long)zz * H + yy) * W + xx;
        s += w[c * 27 + (dz + 1) * 9 + (dy + 1) * 3 + (dx + 1)] * in[rr * ld + (long)cs * c];
      }
    }
  }
  out[r * Cc + c] = s;
}

// ---------------- Depthwise 3x3 conv 2D (HWIO weights), +bias -------------
__global__ void __launch_bounds__(256)
dwconv2d_k(const float* __restrict__ in, const float* __restrict__ w,
           const float* __restrict__ bias, float* __restrict__ out,
           int Nimg, int H, int W, int Cc, long total) {
  long idx = (long)blockIdx.x * 256 + threadIdx.x;
  if (idx >= total) return;
  int c = (int)(idx % Cc); long r = idx / Cc;
  int x = (int)(r % W); int y = (int)((r / W) % H); int n = (int)(r / ((long)W * H));
  float s = bias[c];
  for (int dy = -1; dy <= 1; ++dy) {
    int yy = y + dy; if (yy < 0 || yy >= H) continue;
    for (int dx = -1; dx <= 1; ++dx) {
      int xx = x + dx; if (xx < 0 || xx >= W) continue;
      long rr = ((long)n * H + yy) * W + xx;
      s += w[((dy + 1) * 3 + (dx + 1)) * Cc + c] * in[rr * Cc + c];
    }
  }
  out[r * Cc + c] = s;
}

// ---------------- Elementwise helpers -------------------------------------
__global__ void k_add(const float* a, const float* b, float* o, long n) {
  long i = (long)blockIdx.x * 256 + threadIdx.x; if (i < n) o[i] = a[i] + b[i];
}
__global__ void copy_ch(const float* src, int Cs, float* dst, int Cd, int coff, long total) {
  long i = (long)blockIdx.x * 256 + threadIdx.x; if (i >= total) return;
  int c = (int)(i % Cs); long r = i / Cs;
  dst[r * Cd + coff + c] = src[i];
}
__global__ void win_add_lepe(float* obuf, const float* lepe, int Cb,
                             int D, int H, int W, int wd, int wh, int ww, int L, long total) {
  long i = (long)blockIdx.x * 256 + threadIdx.x; if (i >= total) return;
  int c = (int)(i % Cb); long wt = i / Cb; int t = (int)(wt % L); long win = wt / L;
  obuf[i] += lepe[win_row(win, t, D, H, W, wd, wh, ww) * Cb + c];
}
__global__ void win_scatter(const float* pbuf, float* img, int Cb,
                            int D, int H, int W, int wd, int wh, int ww, int L, long total) {
  long i = (long)blockIdx.x * 256 + threadIdx.x; if (i >= total) return;
  int c = (int)(i % Cb); long wt = i / Cb; int t = (int)(wt % L); long win = wt / L;
  img[win_row(win, t, D, H, W, wd, wh, ww) * Cb + c] = pbuf[i];
}
// x?_px = px_un(qkv[..., cbase:cbase+64] + xbr): three planes (q,k,v) of (D*H2*W2, 256)
__global__ void pxun_k(const float* qkv, int cbase, const float* xbr, float* px,
                       int D, int H2, int W2, long plane, long total) {
  long i = (long)blockIdx.x * 256 + threadIdx.x; if (i >= total) return;
  int j = (int)(i / plane); long rem = i - (long)j * plane;
  long rowh = rem >> 8; int cp = (int)(rem & 255);
  int c = cp >> 2, dy = (cp >> 1) & 1, dx = cp & 1;
  int xh = (int)(rowh % W2); long r1 = rowh / W2;
  int yh = (int)(r1 % H2);  long z = r1 / H2;
  long srow = (z * (2 * H2) + (2 * yh + dy)) * (long)(2 * W2) + (2 * xh + dx);
  px[i] = qkv[srow * 768 + 3 * (cbase + c) + j] + xbr[srow * 64 + c];
}
// att[:, coff:coff+64] = px_sh(srcimg (D*H/2*W/2, 256))
__global__ void pxsh_k(const float* srcimg, float* att, int coff,
                       int D, int H, int W, long total) {
  long i = (long)blockIdx.x * 256 + threadIdx.x; if (i >= total) return;
  int cq = (int)(i & 63); long row = i >> 6;
  int x = (int)(row % W); long r1 = row / W; int y = (int)(r1 % H); long z = r1 / H;
  int yh = y >> 1, dy = y & 1, xh = x >> 1, dx = x & 1;
  long srow = (z * (H / 2) + yh) * (long)(W / 2) + xh;
  att[row * 256 + coff + cq] = srcimg[srow * 256 + 4 * cq + 2 * dy + dx];
}
__global__ void gelu_gate(const float* h, float* o, long total) {
  long i = (long)blockIdx.x * 256 + threadIdx.x; if (i >= total) return;
  int c = (int)(i % 512); long r = i / 512;
  float h1 = h[r * 1024 + c], h2 = h[r * 1024 + 512 + c];
  o[i] = 0.5f * h1 * (1.f + erff(h1 * 0.70710678118654752f)) * h2;
}

// ---------------- Host orchestration --------------------------------------
extern "C" void kernel_launch(void* const* d_in, const int* in_sizes, int n_in,
                              void* d_out, int out_size, void* d_ws, size_t ws_size,
                              hipStream_t stream) {
  (void)in_sizes; (void)n_in; (void)out_size; (void)ws_size;
  const float* X    = (const float*)d_in[0];
  const float* g1   = (const float*)d_in[1];
  const float* b1   = (const float*)d_in[2];
  const float* g2   = (const float*)d_in[3];
  const float* b2   = (const float*)d_in[4];
  const float* Wqkv = (const float*)d_in[5];
  const float* bqkv = (const float*)d_in[6];
  const float *Wp[5], *pb[5], *cw[5], *cb[5];
  for (int i = 0; i < 5; ++i) {
    Wp[i] = (const float*)d_in[7 + 4 * i];  pb[i] = (const float*)d_in[8 + 4 * i];
    cw[i] = (const float*)d_in[9 + 4 * i];  cb[i] = (const float*)d_in[10 + 4 * i];
  }
  const float* Wmi = (const float*)d_in[27]; const float* bmi = (const float*)d_in[28];
  const float* Wdw = (const float*)d_in[29]; const float* bdw = (const float*)d_in[30];
  const float* Wmo = (const float*)d_in[31]; const float* bmo = (const float*)d_in[32];

  const int D = 4, H = 128, W = 128, C = 256;
  const long N = (long)D * H * W;            // 65536 tokens (B=1, pad is no-op)
  const float scale = 0.17677669529663687f;  // (C/NH)^-0.5 = 1/sqrt(32)

  float* ws = (float*)d_ws; size_t off = 0;
  auto alloc = [&](size_t n) { float* p = ws + off; off += n; return p; };
  float* y    = alloc(N * C);
  float* qkv  = alloc(N * 3 * C);
  float* x1   = alloc(N * 64);
  float* x2   = alloc(N * 64);
  float* obuf = alloc(N * C);
  float* pbuf = alloc(N * C);
  float* lepe = alloc(N * C);
  float* att  = alloc(N * C);
  long  Nh    = (long)D * 64 * 64;           // half-res rows
  long  plane = Nh * C;
  float* px   = alloc(3 * plane);
  float* imgh = alloc(Nh * C);
  float* xres = alloc(N * C);
  float* h1   = alloc(N * 1024);
  float* h2   = alloc(N * 1024);
  float* hg   = alloc(N * 512);

  dim3 blk(256);
  auto ew = [&](long n) { return dim3((unsigned)((n + 255) / 256)); };

  // y = ln(ln(x))   (pad was a no-op)
  ln_kernel<<<dim3((unsigned)(N / 8)), blk, 0, stream>>>(X, g1, b1, y, N);
  ln_kernel<<<dim3((unsigned)(N / 8)), blk, 0, stream>>>(y, g1, b1, y, N);
  // qkv = y @ Wqkv + bqkv   (column 3c+j = channel c, j in {q,k,v})
  gemm_wmma<<<dim3(6, (unsigned)(N / 64)), blk, 0, stream>>>(y, Wqkv, bqkv, qkv, (int)N, C, 3 * C);

  auto branch = [&](const float* qb, const float* kb, const float* vb, int ld, int cs,
                    int Di, int Hi, int Wi, int wd, int wh, int wwn, int HD,
                    const float* prW, const float* prb, const float* cwi, const float* cbi,
                    float* ximg) {
    int L = wd * wh * wwn; long Nr = (long)Di * Hi * Wi; long nw = Nr / L; int Cb = 8 * HD;
    dwconv3d_k<<<ew(Nr * Cb), blk, 0, stream>>>(vb, ld, cs, cwi, cbi, lepe, Di, Hi, Wi, Cb, Nr * Cb);
    if (HD == 8)
      attn_flash<8><<<dim3((unsigned)nw, 8), blk, 0, stream>>>(qb, kb, vb, ld, cs, scale,
                                                               Di, Hi, Wi, wd, wh, wwn, L, obuf, Cb);
    else
      attn_flash<32><<<dim3((unsigned)nw, 8), blk, 0, stream>>>(qb, kb, vb, ld, cs, scale,
                                                                Di, Hi, Wi, wd, wh, wwn, L, obuf, Cb);
    win_add_lepe<<<ew(Nr * Cb), blk, 0, stream>>>(obuf, lepe, Cb, Di, Hi, Wi, wd, wh, wwn, L, Nr * Cb);
    gemm_wmma<<<dim3((Cb + 127) / 128, (unsigned)(Nr / 64)), blk, 0, stream>>>(obuf, prW, prb, pbuf, (int)Nr, Cb, Cb);
    win_scatter<<<ew(Nr * Cb), blk, 0, stream>>>(pbuf, ximg, Cb, Di, Hi, Wi, wd, wh, wwn, L, Nr * Cb);
  };

  // x1: channels [0,64), ws (1,128,8); x2: channels [128,192), ws (1,8,128)
  branch(qkv + 0,   qkv + 1,   qkv + 2,   3 * C, 3, D, H, W, 1, 128, 8, 8, Wp[0], pb[0], cw[0], cb[0], x1);
  branch(qkv + 384, qkv + 385, qkv + 386, 3 * C, 3, D, H, W, 1, 8, 128, 8, Wp[1], pb[1], cw[1], cb[1], x2);
  copy_ch<<<ew(N * 64), blk, 0, stream>>>(x1, 64, att, C, 0,   N * 64);
  copy_ch<<<ew(N * 64), blk, 0, stream>>>(x2, 64, att, C, 128, N * 64);

  // x1_out: px_un(qkv[...,64:128]+x1) -> attn ws (1,8,64) hd32 -> px_sh into att[:,64:128]
  pxun_k<<<ew(3 * plane), blk, 0, stream>>>(qkv, 64, x1, px, D, 64, 64, plane, 3 * plane);
  branch(px, px + plane, px + 2 * plane, C, 1, D, 64, 64, 1, 8, 64, 32, Wp[3], pb[3], cw[3], cb[3], imgh);
  pxsh_k<<<ew(N * 64), blk, 0, stream>>>(imgh, att, 64, D, H, W, N * 64);
  // x2_out: px_un(qkv[...,192:256]+x2) -> attn ws (1,64,8) hd32 -> att[:,192:256]
  pxun_k<<<ew(3 * plane), blk, 0, stream>>>(qkv, 192, x2, px, D, 64, 64, plane, 3 * plane);
  branch(px, px + plane, px + 2 * plane, C, 1, D, 64, 64, 1, 64, 8, 32, Wp[4], pb[4], cw[4], cb[4], imgh);
  pxsh_k<<<ew(N * 64), blk, 0, stream>>>(imgh, att, 192, D, H, W, N * 64);

  // qkv2 = att @ Wqkv (overwrite qkv); x3: full C, ws (4,1,1)
  gemm_wmma<<<dim3(6, (unsigned)(N / 64)), blk, 0, stream>>>(att, Wqkv, bqkv, qkv, (int)N, C, 3 * C);
  dwconv3d_k<<<ew(N * C), blk, 0, stream>>>(qkv + 2, 3 * C, 3, cw[2], cb[2], lepe, D, H, W, C, N * C);
  {
    long tot = (N / 4) * 8 * 4;
    attn_small_k<<<ew(tot), blk, 0, stream>>>(qkv + 0, qkv + 1, qkv + 2, 3 * C, 3, scale, D, H, W, obuf, tot);
  }
  win_add_lepe<<<ew(N * C), blk, 0, stream>>>(obuf, lepe, C, D, H, W, 4, 1, 1, 4, N * C);
  gemm_wmma<<<dim3(2, (unsigned)(N / 64)), blk, 0, stream>>>(obuf, Wp[2], pb[2], pbuf, (int)N, C, C);
  win_scatter<<<ew(N * C), blk, 0, stream>>>(pbuf, y, C, D, H, W, 4, 1, 1, 4, N * C);

  // x = shortcut + x3; MLP: ln -> Wmi -> dwconv2d -> gelu-gate -> Wmo -> +x
  k_add<<<ew(N * C), blk, 0, stream>>>(X, y, xres, N * C);
  ln_kernel<<<dim3((unsigned)(N / 8)), blk, 0, stream>>>(xres, g2, b2, att, N);
  gemm_wmma<<<dim3(8, (unsigned)(N / 64)), blk, 0, stream>>>(att, Wmi, bmi, h1, (int)N, C, 1024);
  dwconv2d_k<<<ew(N * 1024), blk, 0, stream>>>(h1, Wdw, bdw, h2, D, H, W, 1024, N * 1024);
  gelu_gate<<<ew(N * 512), blk, 0, stream>>>(h2, hg, N * 512);
  gemm_wmma<<<dim3(2, (unsigned)(N / 64)), blk, 0, stream>>>(hg, Wmo, bmo, pbuf, (int)N, 512, C);
  k_add<<<ew(N * C), blk, 0, stream>>>(xres, pbuf, (float*)d_out, N * C);
}
